// BiDAF_71949292142727
// MI455X (gfx1250) — matrix operand order repaired and checked by reading, not compiled, not run on hardware
//
#include <hip/hip_runtime.h>
#include <hip/hip_bf16.h>
#include <cstdint>
#include <cstddef>

// ---------------------------------------------------------------------------
// BiDAF forward for MI455X (gfx1250, wave32, WMMA).
// Heavy linear algebra via v_wmma_f32_16x16x32_f16; GEMM A-panels staged to
// LDS with GLOBAL_LOAD_ASYNC_TO_LDS_B128 (double-buffered, ASYNCcnt waits).
// ---------------------------------------------------------------------------

typedef __attribute__((ext_vector_type(16))) _Float16 v16h;
typedef __attribute__((ext_vector_type(8)))  _Float16 v8h;
typedef __attribute__((ext_vector_type(8)))  float    v8f;

namespace {
constexpr int kH   = 256;
constexpr int kD2  = 512;      // 2H
constexpr int kG   = 1024;     // 4H (gate width)
constexpr int kB   = 32;
constexpr int kCL  = 512;
constexpr int kQL  = 64;
constexpr int kCCS = 462;
constexpr int kNC  = kB * kCL;        // 16384 context tokens
constexpr int kNQ  = kB * kQL;        //  2048 query tokens
constexpr int kNT  = kNC + kNQ;       // 18432 total tokens
constexpr int kBT  = 16;              // batch tile for LSTM recurrence
}

__device__ __forceinline__ float sigf(float x) { return 1.f / (1.f + __expf(-x)); }

// Async copy 16B from global to LDS (ASYNCcnt-tracked). lds_off is the raw
// 32-bit LDS byte offset (= low 32 bits of the generic shared-memory address).
__device__ __forceinline__ void async_cp_b128(unsigned lds_off, const void* gaddr) {
  asm volatile("global_load_async_to_lds_b128 %0, %1, off"
               :: "v"(lds_off), "v"(gaddr) : "memory");
}
__device__ __forceinline__ void wait_async0() {
  asm volatile("s_wait_asynccnt 0" ::: "memory");
}

// A fragment: 16x32 f16, M on lanes (lane&15), 16 contiguous K per lane (lanes 16-31 -> +16).
__device__ __forceinline__ v16h load_a_frag(const _Float16* base,
                                            int row0, int lda, int k0, int lane) {
  const int lr = lane & 15, lh = lane >> 4;
  const _Float16* p = base + (size_t)(row0 + lr) * lda + k0 + lh * 16;
  v8h lo = *(const v8h*)(p);
  v8h hi = *(const v8h*)(p + 8);
  return __builtin_shufflevector(lo, hi, 0,1,2,3,4,5,6,7,8,9,10,11,12,13,14,15);
}

// B fragment: 32x16 f16 taken from Bt stored (N,K) row-major; N on lanes,
// 16 contiguous K elements per lane (lanes 0-15 -> K k0..k0+15, 16-31 -> +16).
__device__ __forceinline__ v16h load_b_frag(const _Float16* __restrict__ base,
                                            int col0, int ldb, int k0, int lane) {
  const int lr = lane & 15, lh = lane >> 4;
  return *(const v16h*)(base + (size_t)(col0 + lr) * ldb + k0 + lh * 16);
}

// ---------------------------------------------------------------------------
// Generic (optionally batched) WMMA GEMM: C[M,N] = A[M,K] * Bt[N,K]^T + bias.
// Block = 256 threads = 8 waves; block tile 64x64; wave -> 2 adjacent 16x16
// tiles. The 64x32 A panel for the current K-step is staged into LDS with
// async-to-LDS copies, double-buffered: fill k0+32 while WMMAs consume k0.
// Requires exact grids: M%64==0, N%64==0, K%32==0 (true at every call site).
// ---------------------------------------------------------------------------
__global__ void k_gemm(const _Float16* __restrict__ A, const _Float16* __restrict__ Bt,
                       float* __restrict__ C, const float* __restrict__ bias,
                       int M, int N, int K, int lda, int ldb, int ldc,
                       long long sA, long long sB, long long sC) {
  const int bz = blockIdx.z;
  A  += (size_t)bz * sA;
  Bt += (size_t)bz * sB;
  C  += (size_t)bz * sC;
  const int wave = (int)(threadIdx.x >> 5);
  const int lane = (int)(threadIdx.x & 31);
  const int mblk = (int)blockIdx.y * 64;
  const int m0 = mblk + (wave & 3) * 16;          // within-block 16-row tile
  const int n0 = (int)blockIdx.x * 64 + (wave >> 2) * 32;

  __shared__ __align__(32) _Float16 Abuf[2][64 * 32];

  // async-fill mapping: 256 threads x 16B cover one 64x32 f16 panel exactly
  const int frow = (int)threadIdx.x >> 2;         // 0..63
  const int fseg = (int)threadIdx.x & 3;          // 0..3, 8-half segments
  const _Float16* gA = A + (size_t)(mblk + frow) * lda + fseg * 8;
  const unsigned ldsA0 = (unsigned)(size_t)(&Abuf[0][frow * 32 + fseg * 8]);
  const unsigned ldsA1 = (unsigned)(size_t)(&Abuf[1][frow * 32 + fseg * 8]);

  // prologue: stage K-chunk 0 into buffer 0
  async_cp_b128(ldsA0, (const void*)gA);
  wait_async0();
  __syncthreads();

  v8f acc0 = {0.f,0.f,0.f,0.f,0.f,0.f,0.f,0.f};
  v8f acc1 = {0.f,0.f,0.f,0.f,0.f,0.f,0.f,0.f};

  int buf = 0;
  for (int k0 = 0; k0 < K; k0 += 32) {
    if (k0 + 32 < K) {
      async_cp_b128(buf ? ldsA0 : ldsA1, (const void*)(gA + k0 + 32));
      __builtin_prefetch((const void*)(Bt + (size_t)(n0 + (lane & 15)) * ldb + k0 + 32), 0, 1);
    }
    v16h a  = load_a_frag(&Abuf[buf][0], (wave & 3) * 16, 32, 0, lane);
    v16h b0 = load_b_frag(Bt, n0,      ldb, k0, lane);
    v16h b1 = load_b_frag(Bt, n0 + 16, ldb, k0, lane);
    acc0 = __builtin_amdgcn_wmma_f32_16x16x32_f16(false, a, false, b0, (short)0, acc0, false, false);
    acc1 = __builtin_amdgcn_wmma_f32_16x16x32_f16(false, a, false, b1, (short)0, acc1, false, false);
    wait_async0();
    __syncthreads();
    buf ^= 1;
  }

  const int lr = lane & 15, lh = lane >> 4;
  const float bv0 = bias ? bias[n0 + lr]      : 0.f;
  const float bv1 = bias ? bias[n0 + 16 + lr] : 0.f;
#pragma unroll
  for (int j = 0; j < 8; ++j) {
    const int row = m0 + lh * 8 + j;
    C[(size_t)row * ldc + n0 + lr]      = acc0[j] + bv0;
    C[(size_t)row * ldc + n0 + 16 + lr] = acc1[j] + bv1;
  }
}

// ---------------------------------------------------------------------------
// LSTM recurrence (one direction per blockIdx.y, 16 batch rows per blockIdx.x).
// gates = xp[:,t,:] + h @ Whh^T  via WMMA; h double-buffered in LDS (f16),
// cell state held in registers (each wave owns 2 x 16 hidden columns).
// xp layout: (B, S, 1024) token-major. out layout: (B, S, 512), fwd cols 0..255.
// ---------------------------------------------------------------------------
__global__ void k_lstm_rec(const float* __restrict__ xpf, const float* __restrict__ xpb,
                           const _Float16* __restrict__ whfT, const _Float16* __restrict__ whbT,
                           float* __restrict__ out, int S) {
  const int dir  = (int)blockIdx.y;
  const int b0   = (int)blockIdx.x * kBT;
  const int wave = (int)(threadIdx.x >> 5);
  const int lane = (int)(threadIdx.x & 31);
  const int lr = lane & 15, lh = lane >> 4;

  const float*     xp  = dir ? xpb  : xpf;
  const _Float16*  whT = dir ? whbT : whfT;   // (1024, 256) f16 row-major

  __shared__ __align__(32) _Float16 hbuf[2][kBT * kH];

  for (int i = (int)threadIdx.x; i < kBT * kH; i += (int)blockDim.x)
    hbuf[0][i] = (_Float16)0.f;

  float cst[2][8];
#pragma unroll
  for (int u = 0; u < 2; ++u)
#pragma unroll
    for (int j = 0; j < 8; ++j) cst[u][j] = 0.f;

  __syncthreads();

  int cur = 0;
  for (int step = 0; step < S; ++step) {
    const int s = dir ? (S - 1 - step) : step;
    const _Float16* hsrc = hbuf[cur];
    _Float16*       hdst = hbuf[cur ^ 1];

#pragma unroll
    for (int u = 0; u < 2; ++u) {
      const int nb = (wave * 2 + u) * 16;   // hidden column base (0..240)
      v8f acc[4];
      // init accumulators with input projection xp
#pragma unroll
      for (int gi = 0; gi < 4; ++gi) {
        const int col = gi * kH + nb + lr;
#pragma unroll
        for (int j = 0; j < 8; ++j) {
          const int brow = lh * 8 + j;
          acc[gi][j] = xp[((size_t)(b0 + brow) * S + s) * kG + col];
        }
      }
      // gates += h @ Whh^T
#pragma unroll
      for (int k0 = 0; k0 < kH; k0 += 32) {
        v16h a = load_a_frag(hsrc, 0, kH, k0, lane);
#pragma unroll
        for (int gi = 0; gi < 4; ++gi) {
          v16h b = load_b_frag(whT, gi * kH + nb, kH, k0, lane);
          acc[gi] = __builtin_amdgcn_wmma_f32_16x16x32_f16(false, a, false, b,
                                                           (short)0, acc[gi], false, false);
        }
      }
      // elementwise LSTM cell update; D layout: M = lh*8+j, N = lr
#pragma unroll
      for (int j = 0; j < 8; ++j) {
        const float ig = sigf(acc[0][j]);
        const float fg = sigf(acc[1][j]);
        const float gg = tanhf(acc[2][j]);
        const float og = sigf(acc[3][j]);
        const float cn = fg * cst[u][j] + ig * gg;
        const float hv = og * tanhf(cn);
        cst[u][j] = cn;
        const int brow = lh * 8 + j;
        const int col  = nb + lr;
        hdst[brow * kH + col] = (_Float16)hv;
        out[((size_t)(b0 + brow) * S + s) * kD2 + dir * kH + col] = hv;
      }
    }
    __syncthreads();
    cur ^= 1;
  }
}

// ---------------------------------------------------------------------------
// Char CNN feature per token (faithful to torch .view reinterpretation):
// input L[w][d] = char_emb[chars[w]][d]; out[ch] = max_x sum_{i,k} W[ch,i,k]*L[i][x+k]
// written into hx columns [0, 462).
// ---------------------------------------------------------------------------
__global__ void k_char(const float* __restrict__ char_emb,
                       const int* __restrict__ c_char, const int* __restrict__ q_char,
                       const float* __restrict__ conv_w, const float* __restrict__ conv_b,
                       float* __restrict__ hx) {
  const int token = (int)blockIdx.x;
  __shared__ float L[256];
  const int* chars = (token < kNC) ? (c_char + (size_t)token * 16)
                                   : (q_char + (size_t)(token - kNC) * 16);
  const int t = (int)threadIdx.x;
  {
    const int w = t >> 4, d = t & 15;
    L[t] = char_emb[(size_t)chars[w] * 16 + d];
  }
  __syncthreads();
  for (int ch = t; ch < kCCS; ch += 256) {
    const float* W = conv_w + (size_t)ch * 80;
    float best = -1e30f;
    for (int x = 0; x < 12; ++x) {
      float s = 0.f;
#pragma unroll
      for (int i = 0; i < 16; ++i)
#pragma unroll
        for (int k = 0; k < 5; ++k)
          s += W[i * 5 + k] * L[i * 16 + x + k];
      best = fmaxf(best, s);
    }
    hx[(size_t)token * kD2 + ch] = best + conv_b[ch];
  }
}

// word embedding gather into hx columns [462, 512)
__global__ void k_word(const float* __restrict__ word_emb,
                       const int* __restrict__ c_word, const int* __restrict__ q_word,
                       float* __restrict__ hx) {
  const size_t idx = (size_t)blockIdx.x * blockDim.x + threadIdx.x;
  const int token = (int)(idx >> 6);
  const int t = (int)(idx & 63);
  if (token >= kNT || t >= 50) return;
  const int wid = (token < kNC) ? c_word[token] : q_word[token - kNC];
  hx[(size_t)token * kD2 + kCCS + t] = word_emb[(size_t)wid * 50 + t];
}

__global__ void k_f32_to_f16(const float* __restrict__ src, _Float16* __restrict__ dst, size_t n) {
  const size_t i = (size_t)blockIdx.x * blockDim.x + threadIdx.x;
  if (i < n) dst[i] = (_Float16)src[i];
}

// highway gate combine, in place on x
__global__ void k_hw(const float* __restrict__ hl, const float* __restrict__ gl,
                     float* __restrict__ x, size_t n) {
  const size_t i = (size_t)blockIdx.x * blockDim.x + threadIdx.x;
  if (i >= n) return;
  const float g = sigf(gl[i]);
  const float h = fmaxf(hl[i], 0.f);
  x[i] = g * h + (1.f - g) * x[i];
}

// per-row dot product with a vector
__global__ void k_rowdot(const float* __restrict__ X, const float* __restrict__ w,
                         float* __restrict__ out, int dim) {
  const int row = (int)blockIdx.x;
  const float* xr = X + (size_t)row * dim;
  float s = 0.f;
  for (int d = (int)threadIdx.x; d < dim; d += 256) s += xr[d] * w[d];
  __shared__ float red[256];
  red[threadIdx.x] = s;
  __syncthreads();
  for (int o = 128; o > 0; o >>= 1) {
    if ((int)threadIdx.x < o) red[threadIdx.x] += red[threadIdx.x + o];
    __syncthreads();
  }
  if (threadIdx.x == 0) out[row] = red[0];
}

// cw16[b,i,d] = c[b,i,d] * w_cq[d]  (fp16 out, GEMM A operand)
__global__ void k_cw(const float* __restrict__ c, const float* __restrict__ wcq,
                     _Float16* __restrict__ cw16, size_t n) {
  const size_t i = (size_t)blockIdx.x * blockDim.x + threadIdx.x;
  if (i >= n) return;
  const int d = (int)(i & (kD2 - 1));
  cw16[i] = (_Float16)(c[i] * wcq[d]);
}

// finalize s, softmax over Q -> a16; also row max (pre-softmax) -> smax
__global__ void k_attfin(const float* __restrict__ s, const float* __restrict__ cdot,
                         const float* __restrict__ qdot,
                         const float* __restrict__ bc, const float* __restrict__ bq,
                         const float* __restrict__ bcq,
                         _Float16* __restrict__ a16, float* __restrict__ smax) {
  const int row = (int)blockIdx.x;          // b*512 + i
  const int b = row >> 9;
  const int j = (int)threadIdx.x;           // 64 threads
  const float bias = bc[0] + bq[0] + bcq[0];
  const float sv = s[(size_t)row * kQL + j] + cdot[row] + qdot[b * kQL + j] + bias;
  __shared__ float red[64];
  red[j] = sv;
  __syncthreads();
  for (int o = 32; o > 0; o >>= 1) {
    if (j < o) red[j] = fmaxf(red[j], red[j + o]);
    __syncthreads();
  }
  const float mx = red[0];
  __syncthreads();
  const float e = __expf(sv - mx);
  red[j] = e;
  __syncthreads();
  for (int o = 32; o > 0; o >>= 1) {
    if (j < o) red[j] += red[j + o];
    __syncthreads();
  }
  a16[(size_t)row * kQL + j] = (_Float16)(e / red[0]);
  if (j == 0) smax[row] = mx;
}

// q^T per batch, fp16: qT[b, d, j] = q[b, j, d]
__global__ void k_trq(const float* __restrict__ q, _Float16* __restrict__ qT) {
  const size_t i = (size_t)blockIdx.x * blockDim.x + threadIdx.x;   // B*64*512
  if (i >= (size_t)kB * kQL * kD2) return;
  const int b = (int)(i / (kQL * kD2));
  const int r = (int)(i % (kQL * kD2));
  const int j = r / kD2, d = r % kD2;
  qT[((size_t)b * kD2 + d) * kQL + j] = (_Float16)q[i];
}

// q2c[b,d] = sum_i softmax_i(smax[b,i]) * c[b,i,d]
__global__ void k_q2c(const float* __restrict__ c, const float* __restrict__ smax,
                      float* __restrict__ q2c) {
  const int b = (int)blockIdx.x;
  const int t = (int)threadIdx.x;
  __shared__ float batt[kCL];
  __shared__ float red[256];
  float lm = -1e30f;
  for (int i = t; i < kCL; i += 256) lm = fmaxf(lm, smax[b * kCL + i]);
  red[t] = lm;
  __syncthreads();
  for (int o = 128; o > 0; o >>= 1) {
    if (t < o) red[t] = fmaxf(red[t], red[t + o]);
    __syncthreads();
  }
  const float mx = red[0];
  __syncthreads();
  float ls = 0.f;
  for (int i = t; i < kCL; i += 256) {
    const float e = __expf(smax[b * kCL + i] - mx);
    batt[i] = e;
    ls += e;
  }
  red[t] = ls;
  __syncthreads();
  for (int o = 128; o > 0; o >>= 1) {
    if (t < o) red[t] += red[t + o];
    __syncthreads();
  }
  const float inv = 1.f / red[0];
  __syncthreads();
  for (int d = t; d < kD2; d += 256) {
    float acc = 0.f;
    for (int i = 0; i < kCL; ++i)
      acc += batt[i] * c[((size_t)b * kCL + i) * kD2 + d];
    q2c[b * kD2 + d] = acc * inv;
  }
}

// g16[row, :] = [c, c2q, c*c2q, c*q2c]  (fp16, GEMM A operand + p1/p2 source)
__global__ void k_g(const float* __restrict__ c, const float* __restrict__ c2q,
                    const float* __restrict__ q2c, _Float16* __restrict__ g16) {
  const size_t i = (size_t)blockIdx.x * blockDim.x + threadIdx.x;   // NC*512
  if (i >= (size_t)kNC * kD2) return;
  const int d = (int)(i & (kD2 - 1));
  const size_t row = i >> 9;
  const int b = (int)(row >> 9);
  const float cv = c[i], tv = c2q[i], qv = q2c[b * kD2 + d];
  const size_t base = row * (4 * kD2);
  g16[base + d]              = (_Float16)cv;
  g16[base + kD2 + d]        = (_Float16)tv;
  g16[base + 2 * kD2 + d]    = (_Float16)(cv * tv);
  g16[base + 3 * kD2 + d]    = (_Float16)(cv * qv);
}

// span logits: out[row] = g16[row]·gw + gb + m[row]·mw + mb
__global__ void k_span(const _Float16* __restrict__ g16, const float* __restrict__ m,
                       const float* __restrict__ gw, const float* __restrict__ mw,
                       const float* __restrict__ gb, const float* __restrict__ mb,
                       float* __restrict__ out) {
  const int row = (int)blockIdx.x;
  const int t = (int)threadIdx.x;
  float s = 0.f;
  const _Float16* gr = g16 + (size_t)row * (4 * kD2);
  for (int d = t; d < 4 * kD2; d += 256) s += (float)gr[d] * gw[d];
  const float* mr = m + (size_t)row * kD2;
  for (int d = t; d < kD2; d += 256) s += mr[d] * mw[d];
  __shared__ float red[256];
  red[t] = s;
  __syncthreads();
  for (int o = 128; o > 0; o >>= 1) {
    if (t < o) red[t] += red[t + o];
    __syncthreads();
  }
  if (t == 0) out[row] = red[0] + gb[0] + mb[0];
}

// ---------------------------------------------------------------------------
// Host orchestration
// ---------------------------------------------------------------------------
// Flattened input order (depth-first over setup_inputs dict):
//  0 char_emb  1 word_emb  2 conv_w  3 conv_b
//  4 hw_lin_w0 5 hw_lin_b0 6 hw_gate_w0 7 hw_gate_b0
//  8 hw_lin_w1 9 hw_lin_b1 10 hw_gate_w1 11 hw_gate_b1
// 12-17 ctx  {wif, whf, bf, wib, whb, bb}
// 18-23 mod1 {...}   24-29 mod2 {...}   30-35 out {...}
// 36 w_c 37 b_c 38 w_q 39 b_q 40 w_cq 41 b_cq
// 42 p1_g_w 43 p1_g_b 44 p1_m_w 45 p1_m_b 46 p2_g_w 47 p2_g_b 48 p2_m_w 49 p2_m_b
// 50 c_word 51 q_word 52 c_char 53 q_char

extern "C" void kernel_launch(void* const* d_in, const int* in_sizes, int n_in,
                              void* d_out, int out_size, void* d_ws, size_t ws_size,
                              hipStream_t stream) {
  (void)in_sizes; (void)n_in; (void)out_size; (void)ws_size;

  auto F  = [&](int i) { return (const float*)d_in[i]; };
  auto I  = [&](int i) { return (const int*)d_in[i]; };
  float* out = (float*)d_out;

  // ---- workspace bump allocator ----
  char* wsb = (char*)d_ws;
  size_t off = 0;
  auto alloc = [&](size_t bytes) -> char* {
    char* p = wsb + off;
    off = (off + bytes + 255) & ~(size_t)255;
    return p;
  };
  auto cdiv = [](size_t a, size_t b) { return (unsigned)((a + b - 1) / b); };

  float*    hx   = (float*)alloc((size_t)kNT * kD2 * 4);
  _Float16* c16  = (_Float16*)alloc((size_t)kNT * kD2 * 2);   // reusable f16 staging
  float*    t_hl = (float*)alloc((size_t)kNT * kD2 * 4);
  float*    t_gl = (float*)alloc((size_t)kNT * kD2 * 4);
  float*    xpf  = (float*)alloc((size_t)kNT * kG * 4);
  float*    xpb  = (float*)alloc((size_t)kNT * kG * 4);
  float*    enc  = (float*)alloc((size_t)kNT * kD2 * 4);      // ctx output (c then q)
  float*    cdot = (float*)alloc((size_t)kNC * 4);
  float*    qdot = (float*)alloc((size_t)kNQ * 4);
  float*    smax = (float*)alloc((size_t)kNC * 4);
  _Float16* cw16 = (_Float16*)alloc((size_t)kNC * kD2 * 2);
  _Float16* qT16 = (_Float16*)alloc((size_t)kB * kD2 * kQL * 2);
  float*    sbuf = (float*)alloc((size_t)kB * kCL * kQL * 4);
  _Float16* a16  = (_Float16*)alloc((size_t)kB * kCL * kQL * 2);
  float*    c2q  = (float*)alloc((size_t)kNC * kD2 * 4);
  float*    q2c  = (float*)alloc((size_t)kB * kD2 * 4);
  _Float16* g16  = (_Float16*)alloc((size_t)kNC * 4 * kD2 * 2);
  float*    m1   = (float*)alloc((size_t)kNC * kD2 * 4);
  float*    m2   = (float*)alloc((size_t)kNC * kD2 * 4);
  float*    m3   = (float*)alloc((size_t)kNC * kD2 * 4);

  // convert a weight matrix to f16 in workspace
  auto toh = [&](const float* src, size_t n) -> _Float16* {
    _Float16* dst = (_Float16*)alloc(n * 2);
    k_f32_to_f16<<<cdiv(n, 256), 256, 0, stream>>>(src, dst, n);
    return dst;
  };
  auto cvt = [&](const float* src, _Float16* dst, size_t n) {
    k_f32_to_f16<<<cdiv(n, 256), 256, 0, stream>>>(src, dst, n);
  };
  auto gemm = [&](const _Float16* A, const _Float16* Bt, float* C, const float* bias,
                  int M, int N, int K, int lda, int ldb, int ldc,
                  long long sA = 0, long long sB = 0, long long sC = 0, int batch = 1) {
    dim3 g((unsigned)(N / 64), (unsigned)(M / 64), (unsigned)batch);
    k_gemm<<<g, 256, 0, stream>>>(A, Bt, C, bias, M, N, K, lda, ldb, ldc, sA, sB, sC);
  };

  // ---- weight conversions (all Bt = (N,K) row-major already) ----
  _Float16* hwl0 = toh(F(4),  (size_t)kD2 * kD2);
  _Float16* hwg0 = toh(F(6),  (size_t)kD2 * kD2);
  _Float16* hwl1 = toh(F(8),  (size_t)kD2 * kD2);
  _Float16* hwg1 = toh(F(10), (size_t)kD2 * kD2);

  _Float16* ctx_wif = toh(F(12), (size_t)kG * kD2);
  _Float16* ctx_whf = toh(F(13), (size_t)kG * kH);
  _Float16* ctx_wib = toh(F(15), (size_t)kG * kD2);
  _Float16* ctx_whb = toh(F(16), (size_t)kG * kH);

  _Float16* m1_wif = toh(F(18), (size_t)kG * 4 * kD2);
  _Float16* m1_whf = toh(F(19), (size_t)kG * kH);
  _Float16* m1_wib = toh(F(21), (size_t)kG * 4 * kD2);
  _Float16* m1_whb = toh(F(22), (size_t)kG * kH);

  _Float16* m2_wif = toh(F(24), (size_t)kG * kD2);
  _Float16* m2_whf = toh(F(25), (size_t)kG * kH);
  _Float16* m2_wib = toh(F(27), (size_t)kG * kD2);
  _Float16* m2_whb = toh(F(28), (size_t)kG * kH);

  _Float16* o_wif = toh(F(30), (size_t)kG * kD2);
  _Float16* o_whf = toh(F(31), (size_t)kG * kH);
  _Float16* o_wib = toh(F(33), (size_t)kG * kD2);
  _Float16* o_whb = toh(F(34), (size_t)kG * kH);

  // ---- embeddings: char CNN + word gather -> hx (NT, 512) ----
  k_char<<<kNT, 256, 0, stream>>>(F(0), I(52), I(53), F(2), F(3), hx);
  k_word<<<cdiv((size_t)kNT * 64, 256), 256, 0, stream>>>(F(1), I(50), I(51), hx);

  // ---- highway (2 layers) ----
  const _Float16* hwl[2] = {hwl0, hwl1};
  const _Float16* hwg[2] = {hwg0, hwg1};
  const int hwlb[2] = {5, 9}, hwgb[2] = {7, 11};
  for (int i = 0; i < 2; ++i) {
    cvt(hx, c16, (size_t)kNT * kD2);
    gemm(c16, hwl[i], t_hl, F(hwlb[i]), kNT, kD2, kD2, kD2, kD2, kD2);
    gemm(c16, hwg[i], t_gl, F(hwgb[i]), kNT, kD2, kD2, kD2, kD2, kD2);
    k_hw<<<cdiv((size_t)kNT * kD2, 256), 256, 0, stream>>>(t_hl, t_gl, hx, (size_t)kNT * kD2);
  }

  // ---- ctx BiLSTM over c and q ----
  cvt(hx, c16, (size_t)kNT * kD2);
  gemm(c16, ctx_wif, xpf, F(14), kNT, kG, kD2, kD2, kD2, kG);
  gemm(c16, ctx_wib, xpb, F(17), kNT, kG, kD2, kD2, kD2, kG);
  k_lstm_rec<<<dim3(kB / kBT, 2), 256, 0, stream>>>(xpf, xpb, ctx_whf, ctx_whb, enc, kCL);
  k_lstm_rec<<<dim3(kB / kBT, 2), 256, 0, stream>>>(
      xpf + (size_t)kNC * kG, xpb + (size_t)kNC * kG, ctx_whf, ctx_whb,
      enc + (size_t)kNC * kD2, kQL);

  const float* cenc = enc;                         // (B, 512, 512)
  const float* qenc = enc + (size_t)kNC * kD2;     // (B,  64, 512)

  // ---- attention flow ----
  k_rowdot<<<kNC, 256, 0, stream>>>(cenc, F(36), cdot, kD2);
  k_rowdot<<<kNQ, 256, 0, stream>>>(qenc, F(38), qdot, kD2);
  k_cw<<<cdiv((size_t)kNC * kD2, 256), 256, 0, stream>>>(cenc, F(40), cw16, (size_t)kNC * kD2);
  cvt(qenc, c16, (size_t)kNQ * kD2);               // q in f16 (Bt for s-GEMM)
  gemm(cw16, c16, sbuf, nullptr, kCL, kQL, kD2, kD2, kD2, kQL,
       (long long)kCL * kD2, (long long)kQL * kD2, (long long)kCL * kQL, kB);
  k_attfin<<<kNC, 64, 0, stream>>>(sbuf, cdot, qdot, F(37), F(39), F(41), a16, smax);
  k_trq<<<cdiv((size_t)kB * kQL * kD2, 256), 256, 0, stream>>>(qenc, qT16);
  gemm(a16, qT16, c2q, nullptr, kCL, kD2, kQL, kQL, kQL, kD2,
       (long long)kCL * kQL, (long long)kD2 * kQL, (long long)kCL * kD2, kB);
  k_q2c<<<kB, 256, 0, stream>>>(cenc, smax, q2c);
  k_g<<<cdiv((size_t)kNC * kD2, 256), 256, 0, stream>>>(cenc, c2q, q2c, g16);

  // ---- mod1 BiLSTM (input 2048) ----
  gemm(g16, m1_wif, xpf, F(20), kNC, kG, 4 * kD2, 4 * kD2, 4 * kD2, kG);
  gemm(g16, m1_wib, xpb, F(23), kNC, kG, 4 * kD2, 4 * kD2, 4 * kD2, kG);
  k_lstm_rec<<<dim3(kB / kBT, 2), 256, 0, stream>>>(xpf, xpb, m1_whf, m1_whb, m1, kCL);

  // ---- mod2 BiLSTM ----
  cvt(m1, c16, (size_t)kNC * kD2);
  gemm(c16, m2_wif, xpf, F(26), kNC, kG, kD2, kD2, kD2, kG);
  gemm(c16, m2_wib, xpb, F(29), kNC, kG, kD2, kD2, kD2, kG);
  k_lstm_rec<<<dim3(kB / kBT, 2), 256, 0, stream>>>(xpf, xpb, m2_whf, m2_whb, m2, kCL);

  // ---- p1 ----
  k_span<<<kNC, 256, 0, stream>>>(g16, m2, F(42), F(44), F(43), F(45), out);

  // ---- out BiLSTM + p2 ----
  cvt(m2, c16, (size_t)kNC * kD2);
  gemm(c16, o_wif, xpf, F(32), kNC, kG, kD2, kD2, kD2, kG);
  gemm(c16, o_wib, xpb, F(35), kNC, kG, kD2, kD2, kD2, kG);
  k_lstm_rec<<<dim3(kB / kBT, 2), 256, 0, stream>>>(xpf, xpb, o_whf, o_whb, m3, kCL);
  k_span<<<kNC, 256, 0, stream>>>(g16, m3, F(46), F(48), F(47), F(49), out + kNC);
}
